// Attention_40218073759981
// MI455X (gfx1250) — matrix operand order repaired and checked by reading, not compiled, not run on hardware
//
#include <hip/hip_runtime.h>
#include <hip/hip_bf16.h>
#include <math.h>

typedef __bf16 bf16_t;
typedef __attribute__((ext_vector_type(16))) __bf16 v16bf;
typedef __attribute__((ext_vector_type(8)))  __bf16 v8bf;
typedef __attribute__((ext_vector_type(8)))  float  v8f;
typedef int vs4i __attribute__((vector_size(16)));   // matches builtin's int4 pointee

#define WMMA_BF16(a, b, c) \
  __builtin_amdgcn_wmma_f32_16x16x32_bf16(false, (a), false, (b), (short)0, (c), false, false)

#define AS1 __attribute__((address_space(1)))
#define AS3 __attribute__((address_space(3)))

#define ASYNC_LDS_B128(g, l) \
  __builtin_amdgcn_global_load_async_to_lds_b128((AS1 vs4i*)(g), (AS3 vs4i*)(l), 0, 0)

// ---------------------------------------------------------------------------
// fp32 -> bf16 elementwise conversion
// ---------------------------------------------------------------------------
__global__ void cvt_f32_bf16(const float* __restrict__ s, bf16_t* __restrict__ d, int n) {
  int i = blockIdx.x * blockDim.x + threadIdx.x;
  if (i < n) d[i] = (bf16_t)s[i];
}

// ---------------------------------------------------------------------------
// C[M x 256] = A[M x 256] @ W^T  (W stored [256 out, 256 in] row-major)
// One wave computes one 16x16 tile, K walked in 8 steps of 32 (bf16 WMMA).
// mode 0: bf16 out, natural [M x 256], value * scale          (q, k)
// mode 2: bf16 out, transposed out[((b*8+h)*32+d)*2048 + key] (v)
// mode 3: f32 out, sigmoid(value + bias[n])                   (gate)
// mode 4: f32 out, value + bias[n]                            (final)
// ---------------------------------------------------------------------------
__global__ void __launch_bounds__(32)
gemm_wmma(const bf16_t* __restrict__ A, const bf16_t* __restrict__ W,
          const float* __restrict__ bias, void* __restrict__ out,
          int mode, float scale) {
  const int lane = threadIdx.x;
  const int lh = lane >> 4, ln = lane & 15;
  const int ti = blockIdx.x, tj = blockIdx.y;

  const bf16_t* arow = A + (size_t)(ti * 16 + ln) * 256;
  const bf16_t* wrow = W + (size_t)(tj * 16 + ln) * 256;

  v8f acc = {0.f, 0.f, 0.f, 0.f, 0.f, 0.f, 0.f, 0.f};
#pragma unroll
  for (int kk = 0; kk < 256; kk += 32) {
    v8bf lo = *(const v8bf*)(arow + kk + 16 * lh);
    v8bf hi = *(const v8bf*)(arow + kk + 16 * lh + 8);
    v16bf a;
#pragma unroll
    for (int i = 0; i < 8; ++i) { a[i] = lo[i]; a[i + 8] = hi[i]; }
    v16bf b = *(const v16bf*)(wrow + kk + 16 * lh);
    acc = WMMA_BF16(a, b, acc);
  }

  const int n = tj * 16 + ln;
  const float bn = bias ? bias[n] : 0.0f;
#pragma unroll
  for (int r = 0; r < 8; ++r) {
    int m = ti * 16 + r + 8 * lh;
    float v = acc[r] * scale;
    if (mode == 0) {
      ((bf16_t*)out)[(size_t)m * 256 + n] = (bf16_t)v;
    } else if (mode == 2) {
      int bb = m >> 11, key = m & 2047;
      int h = n >> 5, d = n & 31;
      ((bf16_t*)out)[(size_t)((bb * 8 + h) * 32 + d) * 2048 + key] = (bf16_t)v;
    } else if (mode == 3) {
      ((float*)out)[(size_t)m * 256 + n] = 1.0f / (1.0f + __expf(-(v + bn)));
    } else {
      ((float*)out)[(size_t)m * 256 + n] = v + bn;
    }
  }
}

// ---------------------------------------------------------------------------
// Flash attention, split-K inside the workgroup.
// Block = 128 threads = 4 waves; wave w owns keys [w*512, w*512+512) for one
// (b, h, 16-q-row) tile. Per 32-key chunk:
//   - bias tile (16x32 f32, 2KB) streamed into LDS with async global->LDS
//     b128 copies, double buffered (s_wait_asynccnt)
//   - 2 score WMMAs (bias as C operand, read from LDS)
//   - fp32 online softmax, P relayout C->A through LDS
//   - 2 PV WMMAs (V columns contiguous in [B,H,D,K] layout)
// Then a cross-wave flash combine in LDS, gate, bf16 store.
// ---------------------------------------------------------------------------
#define KP 4  // K partitions = waves per block

__global__ void __launch_bounds__(128)
attn_fa(const bf16_t* __restrict__ qb,   // [B,Q,H*D] bf16 (pre-scaled by 1/sqrt(D))
        const bf16_t* __restrict__ kb,   // [B,K,H*D] bf16
        const bf16_t* __restrict__ vt,   // [B,H,D,K] bf16 (transposed)
        const float*  __restrict__ bias, // [B,1,Q,K] f32
        const float*  __restrict__ gate, // [B,Q,H*D] f32 (sigmoid applied)
        bf16_t* __restrict__ ob) {       // [B,Q,H*D] bf16
  const int tid = threadIdx.x;
  const int w = tid >> 5;
  const int lane = tid & 31;
  const int lh = lane >> 4, ln = lane & 15;
  const int qt = blockIdx.x, h = blockIdx.y, b = blockIdx.z;

  __shared__ __align__(128) float  bias_sh[KP][2][16 * 32];  // 16KB, double buffered
  __shared__ __align__(128) bf16_t p_sh[KP][16 * 32];        // 4KB P staging
  __shared__ __align__(128) float  o_sh[KP][16 * 32];        // 8KB partial O
  __shared__ float m_sh[KP][16];
  __shared__ float l_sh[KP][16];

  // Q A-tile (16 q rows x 32 d), loaded once per wave
  const bf16_t* qrow = qb + (size_t)(b * 2048 + qt * 16 + ln) * 256 + h * 32;
  v8bf qlo = *(const v8bf*)(qrow + 16 * lh);
  v8bf qhi = *(const v8bf*)(qrow + 16 * lh + 8);
  v16bf aq;
#pragma unroll
  for (int i = 0; i < 8; ++i) { aq[i] = qlo[i]; aq[i + 8] = qhi[i]; }

  v8f O0 = {0.f, 0.f, 0.f, 0.f, 0.f, 0.f, 0.f, 0.f};
  v8f O1 = {0.f, 0.f, 0.f, 0.f, 0.f, 0.f, 0.f, 0.f};
  float mr[8], lr[8];
#pragma unroll
  for (int r = 0; r < 8; ++r) { mr[r] = -1e30f; lr[r] = 0.0f; }

  // ---- async bias-tile staging (per-lane: one b128 = 16B; 4 ops = 2KB tile) ----
  // chunk tile: rows qt*16..+15, keys kc..kc+31 (128B per row, fully used lines)
  const float* bias_base = bias + (size_t)b * 2048 * 2048 + (size_t)(qt * 16) * 2048;
  const int arow_ = (lane >> 3);      // 0..3  (+4j below)
  const int aseg_ = (lane & 7) * 4;   // float offset within row (16B units)

  const int kbase = w * 512;          // this wave's key partition
  const int NCH = 512 / 32;           // 16 chunks

  // prologue: issue chunk 0 into buffer 0
#pragma unroll
  for (int j = 0; j < 4; ++j) {
    int row = j * 4 + arow_;
    const float* g = bias_base + (size_t)row * 2048 + kbase + aseg_;
    float* l = &bias_sh[w][0][row * 32 + aseg_];
    ASYNC_LDS_B128(g, l);
  }

  for (int it = 0; it < NCH; ++it) {
    const int kc = kbase + it * 32;
    const int cur = it & 1;

    if (it + 1 < NCH) {
      // issue next chunk's bias tile into the other buffer
#pragma unroll
      for (int j = 0; j < 4; ++j) {
        int row = j * 4 + arow_;
        const float* g = bias_base + (size_t)row * 2048 + (kc + 32) + aseg_;
        float* l = &bias_sh[w][cur ^ 1][row * 32 + aseg_];
        ASYNC_LDS_B128(g, l);
      }
      asm volatile("s_wait_asynccnt 0x4" ::: "memory");  // current tile complete
    } else {
      asm volatile("s_wait_asynccnt 0x0" ::: "memory");
    }

    // ---- scores: S = q @ k^T + bias (two 16-key tiles) ----
    v16bf bk0 = *(const v16bf*)(kb + (size_t)(b * 2048 + kc + ln) * 256 + h * 32 + 16 * lh);
    v16bf bk1 = *(const v16bf*)(kb + (size_t)(b * 2048 + kc + 16 + ln) * 256 + h * 32 + 16 * lh);
    const float* bt = &bias_sh[w][cur][0];
    v8f c0, c1;
#pragma unroll
    for (int r = 0; r < 8; ++r) {
      c0[r] = bt[(r + 8 * lh) * 32 + ln];
      c1[r] = bt[(r + 8 * lh) * 32 + 16 + ln];
    }
    v8f s0 = WMMA_BF16(aq, bk0, c0);
    v8f s1 = WMMA_BF16(aq, bk1, c1);

    // ---- online softmax (row reductions across 16-lane groups) ----
#pragma unroll
    for (int r = 0; r < 8; ++r) {
      float v = fmaxf(s0[r], s1[r]);
#pragma unroll
      for (int off = 1; off < 16; off <<= 1) v = fmaxf(v, __shfl_xor(v, off, 32));
      float mn = fmaxf(mr[r], v);
      float al = __expf(mr[r] - mn);
      mr[r] = mn;
      float p0 = __expf(s0[r] - mn);
      float p1 = __expf(s1[r] - mn);
      s0[r] = p0; s1[r] = p1;
      float rs = p0 + p1;
#pragma unroll
      for (int off = 1; off < 16; off <<= 1) rs += __shfl_xor(rs, off, 32);
      lr[r] = lr[r] * al + rs;
      O0[r] *= al;
      O1[r] *= al;
    }

    // ---- relayout P (C layout f32 -> A layout bf16) through LDS ----
#pragma unroll
    for (int r = 0; r < 8; ++r) {
      int m = r + 8 * lh;
      p_sh[w][m * 32 + ln] = (bf16_t)s0[r];
      p_sh[w][m * 32 + 16 + ln] = (bf16_t)s1[r];
    }
    asm volatile("s_wait_dscnt 0x0" ::: "memory");  // same-wave LDS RAW (also fences
                                                    // bias reads before buffer reuse)
    v8bf plo = *(const v8bf*)(&p_sh[w][0] + ln * 32 + 16 * lh);
    v8bf phi = *(const v8bf*)(&p_sh[w][0] + ln * 32 + 16 * lh + 8);
    v16bf ap;
#pragma unroll
    for (int i = 0; i < 8; ++i) { ap[i] = plo[i]; ap[i + 8] = phi[i]; }

    // ---- O += P @ V ----
    const bf16_t* vbase = vt + (size_t)((b * 8 + h) * 32) * 2048 + kc + 16 * lh;
    v16bf bv0 = *(const v16bf*)(vbase + (size_t)ln * 2048);
    v16bf bv1 = *(const v16bf*)(vbase + (size_t)(16 + ln) * 2048);
    O0 = WMMA_BF16(ap, bv0, O0);
    O1 = WMMA_BF16(ap, bv1, O1);
  }

  // ---- publish this wave's partial (unnormalized O, m, l) ----
#pragma unroll
  for (int r = 0; r < 8; ++r) {
    int m = r + 8 * lh;
    o_sh[w][m * 32 + ln] = O0[r];
    o_sh[w][m * 32 + 16 + ln] = O1[r];
    if (ln == 0) { m_sh[w][m] = mr[r]; l_sh[w][m] = lr[r]; }
  }
  __syncthreads();

  // ---- cross-wave flash combine + gate + store (512 elems / 128 threads) ----
#pragma unroll
  for (int i = 0; i < 4; ++i) {
    int idx = i * 128 + tid;
    int row = idx >> 5, col = idx & 31;
    float mstar = m_sh[0][row];
#pragma unroll
    for (int w2 = 1; w2 < KP; ++w2) mstar = fmaxf(mstar, m_sh[w2][row]);
    float denom = 0.f, val = 0.f;
#pragma unroll
    for (int w2 = 0; w2 < KP; ++w2) {
      float e = __expf(m_sh[w2][row] - mstar);
      denom += l_sh[w2][row] * e;
      val += o_sh[w2][row * 32 + col] * e;
    }
    size_t g = (size_t)(b * 2048 + qt * 16 + row) * 256 + h * 32 + col;
    float res = (val / denom) * gate[g];
    ob[g] = (bf16_t)res;
  }
}

// ---------------------------------------------------------------------------
extern "C" void kernel_launch(void* const* d_in, const int* in_sizes, int n_in,
                              void* d_out, int out_size, void* d_ws, size_t ws_size,
                              hipStream_t stream) {
  (void)in_sizes; (void)n_in; (void)out_size; (void)ws_size;
  const float* q_x  = (const float*)d_in[0];
  const float* kv_x = (const float*)d_in[1];
  const float* bias = (const float*)d_in[2];
  const float* Wq   = (const float*)d_in[3];
  const float* Wk   = (const float*)d_in[4];
  const float* Wv   = (const float*)d_in[5];
  const float* Wo   = (const float*)d_in[6];
  const float* bo   = (const float*)d_in[7];
  const float* Wg   = (const float*)d_in[8];
  const float* bg   = (const float*)d_in[9];
  float* out = (float*)d_out;

  char* ws = (char*)d_ws;
  bf16_t* qx_bf  = (bf16_t*)(ws + 0);        // 2MB
  bf16_t* kvx_bf = (bf16_t*)(ws + 2097152);  // 2MB
  bf16_t* wq_bf  = (bf16_t*)(ws + 4194304);  // 128KB each
  bf16_t* wk_bf  = (bf16_t*)(ws + 4325376);
  bf16_t* wv_bf  = (bf16_t*)(ws + 4456448);
  bf16_t* wg_bf  = (bf16_t*)(ws + 4587520);
  bf16_t* wo_bf  = (bf16_t*)(ws + 4718592);
  bf16_t* q_bf   = (bf16_t*)(ws + 4849664);  // [B,Q,256] bf16, 2MB
  bf16_t* k_bf   = (bf16_t*)(ws + 6946816);  // [B,K,256] bf16, 2MB
  bf16_t* vt_bf  = (bf16_t*)(ws + 9043968);  // [B,H,32,2048] bf16, 2MB
  float*  g_f32  = (float*)(ws + 11141120);  // [B,Q,256] f32, 4MB
  bf16_t* o_bf   = (bf16_t*)(ws + 15335424); // [B,Q,256] bf16, 2MB

  const int nAct = 4096 * 256;
  const int nWt  = 256 * 256;
  const float inv_sqrt_d = 0.1767766952966369f;  // 1/sqrt(32)

  cvt_f32_bf16<<<(nAct + 255) / 256, 256, 0, stream>>>(q_x, qx_bf, nAct);
  cvt_f32_bf16<<<(nAct + 255) / 256, 256, 0, stream>>>(kv_x, kvx_bf, nAct);
  cvt_f32_bf16<<<(nWt + 255) / 256, 256, 0, stream>>>(Wq, wq_bf, nWt);
  cvt_f32_bf16<<<(nWt + 255) / 256, 256, 0, stream>>>(Wk, wk_bf, nWt);
  cvt_f32_bf16<<<(nWt + 255) / 256, 256, 0, stream>>>(Wv, wv_bf, nWt);
  cvt_f32_bf16<<<(nWt + 255) / 256, 256, 0, stream>>>(Wg, wg_bf, nWt);
  cvt_f32_bf16<<<(nWt + 255) / 256, 256, 0, stream>>>(Wo, wo_bf, nWt);

  dim3 ggrid(256, 16);
  gemm_wmma<<<ggrid, 32, 0, stream>>>(qx_bf,  wq_bf, nullptr, q_bf,  0, inv_sqrt_d);
  gemm_wmma<<<ggrid, 32, 0, stream>>>(kvx_bf, wk_bf, nullptr, k_bf,  0, 1.0f);
  gemm_wmma<<<ggrid, 32, 0, stream>>>(kvx_bf, wv_bf, nullptr, vt_bf, 2, 1.0f);
  gemm_wmma<<<ggrid, 32, 0, stream>>>(qx_bf,  wg_bf, bg,      g_f32, 3, 1.0f);

  // 2048 blocks x 4 waves (split-K inside the block) = 8192 waves
  dim3 agrid(128, 8, 2);
  attn_fa<<<agrid, 128, 0, stream>>>(q_bf, k_bf, vt_bf, bias, g_f32, o_bf);

  gemm_wmma<<<ggrid, 32, 0, stream>>>(o_bf, wo_bf, bo, out, 4, 1.0f);
}
